// Restormer_66305705115698
// MI455X (gfx1250) — matrix operand order repaired and checked by reading, not compiled, not run on hardware
//
#include <hip/hip_runtime.h>
#include <cstdint>
#include <cstddef>

// ---------------------------------------------------------------------------
// Types
// ---------------------------------------------------------------------------
typedef __attribute__((ext_vector_type(16))) _Float16     v16h;
typedef __attribute__((ext_vector_type(8)))  float        v8f;
typedef __attribute__((ext_vector_type(4)))  unsigned int v4u;
typedef __attribute__((ext_vector_type(8)))  int          v8i;
typedef __attribute__((ext_vector_type(4)))  int          v4i;

static inline unsigned cdiv_u(long long a, long long b) { return (unsigned)((a + b - 1) / b); }

#if __has_builtin(__builtin_amdgcn_tensor_load_to_lds) && __has_builtin(__builtin_amdgcn_s_wait_tensorcnt)
#define ATHENA_TDM 1
#else
#define ATHENA_TDM 0
#endif

#if ATHENA_TDM
// Issue one 2D-tile TDM load: tile [t1 rows x t0 contiguous elems] of fp32,
// row stride = strideElems, into LDS byte offset ldsOff. OOB rows/cols
// zero-fill via tensor_dim0/1. Per-wave instruction (EXEC ignored).
__device__ __forceinline__ void tdm_load2d_f32(
    unsigned ldsOff, const float* gaddr, unsigned td0, unsigned td1,
    unsigned t0, unsigned t1, unsigned long long strideElems)
{
  unsigned long long ga = (unsigned long long)(uintptr_t)gaddr;
  v4u g0;
  g0.x = 1u;                                          // [1:0] descriptor count = 1
  g0.y = ldsOff;                                      // [63:32] LDS byte address
  g0.z = (unsigned)(ga & 0xffffffffull);              // global addr lo
  g0.w = (unsigned)((ga >> 32) & 0x1ffffffull) | (2u << 30);  // addr hi | type=2
  v8i g1;
  g1[0] = (int)(2u << 16);                            // data_size = 4 bytes
  g1[1] = (int)((td0 & 0xffffu) << 16);               // tensor_dim0[15:0]
  g1[2] = (int)((td0 >> 16) | ((td1 & 0xffffu) << 16));
  g1[3] = (int)((td1 >> 16) | ((t0 & 0xffffu) << 16)); // tile_dim0
  g1[4] = (int)(t1 & 0xffffu);                        // tile_dim1 (tile_dim2 = 0)
  g1[5] = (int)(unsigned)(strideElems & 0xffffffffull);
  g1[6] = (int)(unsigned)((strideElems >> 32) & 0xffffull);
  g1[7] = 0;
  v4i z4 = {0, 0, 0, 0};
  v8i z8 = {0, 0, 0, 0, 0, 0, 0, 0};
  __builtin_amdgcn_tensor_load_to_lds(g0, g1, z4, z4, z8, 0);
}
#endif

// ---------------------------------------------------------------------------
// WMMA GEMM:  C[M,N] = A[M,K](row major, lda=K) * B  (+ Res), f16 WMMA, f32 acc
//   BT=false: B[k*sBk + n]        (row-major, sBn==1)
//   BT=true : B[k + n*sBn]        (transposed access, sBk==1)
//   Batched over blockIdx.z with strides sAb/sBb/sCb (Res shares sCb).
//   Block = 256 threads (8 wave32 as 2Mx4N), block tile 32(M) x 128(N),
//   each wave owns two 16x16 accumulators. K step 32.
//   A tile (32x32 f32) and B tile (32x128 f32) staged into LDS by the TDM.
// ---------------------------------------------------------------------------
template <bool BT>
__global__ __launch_bounds__(256) void k_gemm_wmma(
    const float* __restrict__ A, const float* __restrict__ B,
    const float* __restrict__ Res, float* __restrict__ C,
    int M, int N, int K,
    long long sAb, long long sBb, long long sCb, long long sBs)
{
  // tileA at LDS byte 0 (32x32 f32), tileB at byte 4096:
  //   BT=false -> [k][n] rows of 128;  BT=true -> [n][k] rows of 32
  __shared__ float lds[32 * 32 + 32 * 128];
  float* tileA = lds;
  float* tileB = lds + 32 * 32;

  const int bz = blockIdx.z;
  const float* Ab = A + (long long)bz * sAb;
  const float* Bb = B + (long long)bz * sBb;
  const float* Rb = Res ? Res + (long long)bz * sCb : (const float*)0;
  float*       Cb = C + (long long)bz * sCb;

  const int wid  = threadIdx.x >> 5;
  const int lane = threadIdx.x & 31;
  const int l16  = lane & 15;
  const int lhi  = lane >> 4;                 // lane half: 0 | 1
  const int m0b  = blockIdx.y * 32;
  const int n0b  = blockIdx.x * 128;
  const int mrow = 16 * (wid >> 2);           // wave's M offset inside block tile
  const int nloc0 = 32 * (wid & 3);           // wave's first N subtile in block tile
  const int nloc1 = nloc0 + 16;

  v8f acc0 = {}, acc1 = {};

  for (int k0 = 0; k0 < K; k0 += 32) {
    __syncthreads();
#if ATHENA_TDM
    if (threadIdx.x < 32) {
      // A tile: 32 rows (m) x 32 contiguous (k), stride K
      tdm_load2d_f32(0u, Ab + (long long)m0b * K + k0,
                     (unsigned)(K - k0), (unsigned)(M - m0b), 32u, 32u,
                     (unsigned long long)(unsigned)K);
      if (!BT) {
        // B tile: 32 rows (k) x 128 contiguous (n), stride sBs (= row stride)
        tdm_load2d_f32(4096u, Bb + (long long)k0 * sBs + n0b,
                       (unsigned)(N - n0b), (unsigned)(K - k0), 128u, 32u,
                       (unsigned long long)sBs);
      } else {
        // B tile: 128 rows (n) x 32 contiguous (k), stride sBs (= column stride)
        tdm_load2d_f32(4096u, Bb + (long long)n0b * sBs + k0,
                       (unsigned)(K - k0), (unsigned)(N - n0b), 32u, 128u,
                       (unsigned long long)sBs);
      }
      __builtin_amdgcn_s_wait_tensorcnt(0);
    }
#else
    for (int i = threadIdx.x; i < 32 * 32; i += 256) {
      int r = i >> 5, cc = i & 31;
      int gm = m0b + r, gk = k0 + cc;
      tileA[i] = (gm < M && gk < K) ? Ab[(long long)gm * K + gk] : 0.0f;
    }
    for (int i = threadIdx.x; i < 32 * 128; i += 256) {
      float v = 0.0f;
      if (!BT) {
        int kk = i >> 7, nn = i & 127;
        if ((k0 + kk) < K && (n0b + nn) < N)
          v = Bb[(long long)(k0 + kk) * sBs + (n0b + nn)];
      } else {
        int nn = i >> 5, kk = i & 31;
        if ((k0 + kk) < K && (n0b + nn) < N)
          v = Bb[(long long)(n0b + nn) * sBs + (k0 + kk)];
      }
      tileB[i] = v;
    }
#endif
    __syncthreads();

    // ---- A fragment (16x32 f16): lanes 0-15 K 0..15, lanes 16-31 K 16..31
    const float* ar = &tileA[(mrow + l16) * 32 + lhi * 16];
    float4 a0 = *(const float4*)(ar);
    float4 a1 = *(const float4*)(ar + 4);
    float4 a2 = *(const float4*)(ar + 8);
    float4 a3 = *(const float4*)(ar + 12);
    v16h af;
    af[0]  = (_Float16)a0.x; af[1]  = (_Float16)a0.y; af[2]  = (_Float16)a0.z; af[3]  = (_Float16)a0.w;
    af[4]  = (_Float16)a1.x; af[5]  = (_Float16)a1.y; af[6]  = (_Float16)a1.z; af[7]  = (_Float16)a1.w;
    af[8]  = (_Float16)a2.x; af[9]  = (_Float16)a2.y; af[10] = (_Float16)a2.z; af[11] = (_Float16)a2.w;
    af[12] = (_Float16)a3.x; af[13] = (_Float16)a3.y; af[14] = (_Float16)a3.z; af[15] = (_Float16)a3.w;

    // ---- B fragments (32x16 f16): lanes 0-15 K 0..15, lanes 16-31 K 16..31
    v16h bf0, bf1;
    if (!BT) {
      const int kb = lhi * 16;
#pragma unroll
      for (int e = 0; e < 16; ++e) {
        bf0[e] = (_Float16)tileB[(kb + e) * 128 + nloc0 + l16];
        bf1[e] = (_Float16)tileB[(kb + e) * 128 + nloc1 + l16];
      }
    } else {
      const float* b0 = &tileB[(nloc0 + l16) * 32 + lhi * 16];
      const float* b1 = &tileB[(nloc1 + l16) * 32 + lhi * 16];
      float4 q0 = *(const float4*)(b0);
      float4 q1 = *(const float4*)(b0 + 4);
      float4 q2 = *(const float4*)(b0 + 8);
      float4 q3 = *(const float4*)(b0 + 12);
      bf0[0]  = (_Float16)q0.x; bf0[1]  = (_Float16)q0.y; bf0[2]  = (_Float16)q0.z; bf0[3]  = (_Float16)q0.w;
      bf0[4]  = (_Float16)q1.x; bf0[5]  = (_Float16)q1.y; bf0[6]  = (_Float16)q1.z; bf0[7]  = (_Float16)q1.w;
      bf0[8]  = (_Float16)q2.x; bf0[9]  = (_Float16)q2.y; bf0[10] = (_Float16)q2.z; bf0[11] = (_Float16)q2.w;
      bf0[12] = (_Float16)q3.x; bf0[13] = (_Float16)q3.y; bf0[14] = (_Float16)q3.z; bf0[15] = (_Float16)q3.w;
      float4 r0 = *(const float4*)(b1);
      float4 r1 = *(const float4*)(b1 + 4);
      float4 r2 = *(const float4*)(b1 + 8);
      float4 r3 = *(const float4*)(b1 + 12);
      bf1[0]  = (_Float16)r0.x; bf1[1]  = (_Float16)r0.y; bf1[2]  = (_Float16)r0.z; bf1[3]  = (_Float16)r0.w;
      bf1[4]  = (_Float16)r1.x; bf1[5]  = (_Float16)r1.y; bf1[6]  = (_Float16)r1.z; bf1[7]  = (_Float16)r1.w;
      bf1[8]  = (_Float16)r2.x; bf1[9]  = (_Float16)r2.y; bf1[10] = (_Float16)r2.z; bf1[11] = (_Float16)r2.w;
      bf1[12] = (_Float16)r3.x; bf1[13] = (_Float16)r3.y; bf1[14] = (_Float16)r3.z; bf1[15] = (_Float16)r3.w;
    }

    acc0 = __builtin_amdgcn_wmma_f32_16x16x32_f16(false, af, false, bf0,
                                                  (short)0, acc0, false, false);
    acc1 = __builtin_amdgcn_wmma_f32_16x16x32_f16(false, af, false, bf1,
                                                  (short)0, acc1, false, false);
  }

  // C/D layout: VGPR r -> M = r (lanes 0-15) or 8+r (lanes 16-31), N = lane&15
#pragma unroll
  for (int r = 0; r < 8; ++r) {
    int gm = m0b + mrow + r + lhi * 8;
    if (gm < M) {
      int gn0 = n0b + nloc0 + l16;
      int gn1 = n0b + nloc1 + l16;
      if (gn0 < N) {
        long long idx = (long long)gm * N + gn0;
        float v = acc0[r];
        if (Rb) v += Rb[idx];
        Cb[idx] = v;
      }
      if (gn1 < N) {
        long long idx = (long long)gm * N + gn1;
        float v = acc1[r];
        if (Rb) v += Rb[idx];
        Cb[idx] = v;
      }
    }
  }
}

// ---------------------------------------------------------------------------
// LayerNorm over channel dim of [C, HW]; one thread per spatial position.
// ---------------------------------------------------------------------------
__global__ __launch_bounds__(256) void k_layernorm(
    const float* __restrict__ x, const float* __restrict__ w,
    const float* __restrict__ b, float* __restrict__ y, int C, int HW)
{
  int p = blockIdx.x * 256 + threadIdx.x;
  if (p >= HW) return;
  float mu = 0.0f;
  for (int c = 0; c < C; ++c) mu += x[(long long)c * HW + p];
  mu /= (float)C;
  float var = 0.0f;
  for (int c = 0; c < C; ++c) {
    float d = x[(long long)c * HW + p] - mu;
    var += d * d;
  }
  var /= (float)C;
  float inv = rsqrtf(var + 1e-5f);
  for (int c = 0; c < C; ++c) {
    long long i = (long long)c * HW + p;
    y[i] = (x[i] - mu) * inv * w[c] + b[c];
  }
}

// ---------------------------------------------------------------------------
// L2-normalize each row of length n (in place); one 256-thread block per row.
// ---------------------------------------------------------------------------
__global__ __launch_bounds__(256) void k_l2norm_rows(float* __restrict__ t, int n)
{
  __shared__ float red[256];
  float* row = t + (long long)blockIdx.x * n;
  float s = 0.0f;
  for (int i = threadIdx.x; i < n; i += 256) { float v = row[i]; s += v * v; }
  red[threadIdx.x] = s;
  __syncthreads();
  for (int st = 128; st > 0; st >>= 1) {
    if (threadIdx.x < st) red[threadIdx.x] += red[threadIdx.x + st];
    __syncthreads();
  }
  float inv = 1.0f / fmaxf(sqrtf(red[0]), 1e-12f);
  for (int i = threadIdx.x; i < n; i += 256) row[i] *= inv;
}

// ---------------------------------------------------------------------------
// Softmax over rows of 48 (attn logits), with per-head temperature.
// grid = heads*48 rows, block = 64.
// ---------------------------------------------------------------------------
__global__ __launch_bounds__(64) void k_softmax48(float* __restrict__ attn,
                                                  const float* __restrict__ temp)
{
  int row = blockIdx.x;
  int h = row / 48;
  float* a = attn + (long long)row * 48;
  __shared__ float buf[48];
  __shared__ float mx, sum;
  int tid = threadIdx.x;
  float t = temp[h];
  if (tid < 48) buf[tid] = a[tid] * t;
  __syncthreads();
  if (tid == 0) {
    float m = -1e30f;
    for (int i = 0; i < 48; ++i) m = fmaxf(m, buf[i]);
    mx = m;
  }
  __syncthreads();
  if (tid < 48) buf[tid] = __expf(buf[tid] - mx);
  __syncthreads();
  if (tid == 0) {
    float s = 0.0f;
    for (int i = 0; i < 48; ++i) s += buf[i];
    sum = s;
  }
  __syncthreads();
  if (tid < 48) a[tid] = buf[tid] / sum;
}

// ---------------------------------------------------------------------------
// Depthwise 3x3 (pad 1), NCHW, w[C,1,3,3]
// ---------------------------------------------------------------------------
__global__ __launch_bounds__(256) void k_dwconv3(
    const float* __restrict__ x, const float* __restrict__ w,
    float* __restrict__ y, int C, int H, int W)
{
  long long idx = (long long)blockIdx.x * 256 + threadIdx.x;
  long long tot = (long long)C * H * W;
  if (idx >= tot) return;
  int px = (int)(idx % W);
  int py = (int)((idx / W) % H);
  int c  = (int)(idx / ((long long)W * H));
  const float* wc = w + (long long)c * 9;
  float s = 0.0f;
  for (int dy = 0; dy < 3; ++dy) {
    int iy = py + dy - 1;
    if (iy < 0 || iy >= H) continue;
    for (int dx = 0; dx < 3; ++dx) {
      int ix = px + dx - 1;
      if (ix < 0 || ix >= W) continue;
      s += x[((long long)c * H + iy) * W + ix] * wc[dy * 3 + dx];
    }
  }
  y[idx] = s;
}

// ---------------------------------------------------------------------------
// Dense 3x3 conv (pad 1), small Ci/Co (embed/down/up/out paths)
// ---------------------------------------------------------------------------
__global__ __launch_bounds__(256) void k_conv3x3(
    const float* __restrict__ x, const float* __restrict__ w,
    float* __restrict__ y, int Ci, int Co, int H, int W)
{
  long long idx = (long long)blockIdx.x * 256 + threadIdx.x;
  long long tot = (long long)Co * H * W;
  if (idx >= tot) return;
  int px = (int)(idx % W);
  int py = (int)((idx / W) % H);
  int co = (int)(idx / ((long long)W * H));
  float s = 0.0f;
  for (int ci = 0; ci < Ci; ++ci) {
    const float* wc = w + ((long long)co * Ci + ci) * 9;
    const float* xc = x + (long long)ci * H * W;
    for (int dy = 0; dy < 3; ++dy) {
      int iy = py + dy - 1;
      if (iy < 0 || iy >= H) continue;
      for (int dx = 0; dx < 3; ++dx) {
        int ix = px + dx - 1;
        if (ix < 0 || ix >= W) continue;
        s += xc[(long long)iy * W + ix] * wc[dy * 3 + dx];
      }
    }
  }
  y[idx] = s;
}

// ---------------------------------------------------------------------------
// GDFN gate: g[i] = gelu_exact(y[i]) * y[i + off]
// ---------------------------------------------------------------------------
__global__ __launch_bounds__(256) void k_gelu_gate(
    const float* __restrict__ y, float* __restrict__ g, long long n, long long off)
{
  long long i = (long long)blockIdx.x * 256 + threadIdx.x;
  if (i >= n) return;
  float a = y[i];
  float b = y[i + off];
  g[i] = 0.5f * a * (1.0f + erff(a * 0.70710678118654752f)) * b;
}

__global__ __launch_bounds__(256) void k_add(
    const float* __restrict__ a, const float* __restrict__ b,
    float* __restrict__ o, long long n)
{
  long long i = (long long)blockIdx.x * 256 + threadIdx.x;
  if (i < n) o[i] = a[i] + b[i];
}

// pixel_unshuffle2: [C,H,W] -> [4C,H/2,W/2]
__global__ __launch_bounds__(256) void k_unshuffle2(
    const float* __restrict__ x, float* __restrict__ y, int C, int H, int W)
{
  int Ho = H >> 1, Wo = W >> 1;
  long long tot = (long long)C * H * W;
  long long idx = (long long)blockIdx.x * 256 + threadIdx.x;
  if (idx >= tot) return;
  int ox = (int)(idx % Wo);
  int oy = (int)((idx / Wo) % Ho);
  int oc = (int)(idx / ((long long)Wo * Ho));  // 0..4C-1
  int c  = oc >> 2;
  int sy = (oc >> 1) & 1;
  int sx = oc & 1;
  y[idx] = x[((long long)c * H + (2 * oy + sy)) * W + (2 * ox + sx)];
}

// pixel_shuffle2: [C,H,W] -> [C/4,2H,2W]
__global__ __launch_bounds__(256) void k_shuffle2(
    const float* __restrict__ x, float* __restrict__ y, int C, int H, int W)
{
  int Ho = H << 1, Wo = W << 1;
  long long tot = (long long)C * H * W;
  long long idx = (long long)blockIdx.x * 256 + threadIdx.x;
  if (idx >= tot) return;
  int ox = (int)(idx % Wo);
  int oy = (int)((idx / Wo) % Ho);
  int oc = (int)(idx / ((long long)Wo * Ho));  // 0..C/4-1
  int ic = oc * 4 + (oy & 1) * 2 + (ox & 1);
  y[idx] = x[((long long)ic * H + (oy >> 1)) * W + (ox >> 1)];
}

// ---------------------------------------------------------------------------
// Host-side orchestration
// ---------------------------------------------------------------------------
struct BlockP {
  const float *proj, *qkv, *qkv_dw, *temp, *ffn_dw, *pin, *pout;
  const float *ln1b, *ln1w, *ln2b, *ln2w;
  int c, heads, hid;
};

struct Bump {
  char* base;
  size_t off;
  float* alloc(size_t nElems) {
    float* p = (float*)(base + off);
    off = (off + nElems * sizeof(float) + 255) & ~(size_t)255;
    return p;
  }
};

// bT=false: B row-major [K,N] with row stride sBs; bT=true: B[k + n*sBs]
static void gemm(hipStream_t s, const float* A, const float* B, const float* Res,
                 float* C, int M, int N, int K, int batch,
                 long long sAb, long long sBb, long long sCb,
                 long long sBs, bool bT)
{
  dim3 g(cdiv_u(N, 128), cdiv_u(M, 32), (unsigned)batch);
  if (bT)
    k_gemm_wmma<true><<<g, 256, 0, s>>>(A, B, Res, C, M, N, K, sAb, sBb, sCb, sBs);
  else
    k_gemm_wmma<false><<<g, 256, 0, s>>>(A, B, Res, C, M, N, K, sAb, sBb, sCb, sBs);
}

static void dwconv(hipStream_t s, const float* x, const float* w, float* y,
                   int C, int H, int W)
{
  long long tot = (long long)C * H * W;
  k_dwconv3<<<cdiv_u(tot, 256), 256, 0, s>>>(x, w, y, C, H, W);
}

static void conv3x3(hipStream_t s, const float* x, const float* w, float* y,
                    int Ci, int Co, int H, int W)
{
  long long tot = (long long)Co * H * W;
  k_conv3x3<<<cdiv_u(tot, 256), 256, 0, s>>>(x, w, y, Ci, Co, H, W);
}

// One Restormer transformer block, in place on x[C,H*W]
static void tblock(hipStream_t s, Bump& ws, const BlockP& p, float* x, int H, int W)
{
  const int C = p.c, HW = H * W, heads = p.heads, hid = p.hid;
  const size_t save = ws.off;

  // ---- MDTA ----
  float* ln = ws.alloc((size_t)C * HW);
  k_layernorm<<<cdiv_u(HW, 256), 256, 0, s>>>(x, p.ln1w, p.ln1b, ln, C, HW);

  float* q1 = ws.alloc((size_t)3 * C * HW);
  float* q2 = ws.alloc((size_t)3 * C * HW);
  gemm(s, p.qkv, ln, nullptr, q1, 3 * C, HW, C, 1, 0, 0, 0, HW, false);
  dwconv(s, q1, p.qkv_dw, q2, 3 * C, H, W);
  k_l2norm_rows<<<(unsigned)(2 * C), 256, 0, s>>>(q2, HW);  // normalize q and k rows

  const float* q = q2;
  const float* k = q2 + (size_t)C * HW;
  const float* v = q2 + (size_t)2 * C * HW;

  float* attn = ws.alloc((size_t)heads * 48 * 48);
  // logits[h] = q_h (48 x HW) * k_h^T  -> transposed B access, column stride HW
  gemm(s, q, k, nullptr, attn, 48, 48, HW, heads,
       (long long)48 * HW, (long long)48 * HW, 48 * 48, HW, true);
  k_softmax48<<<(unsigned)(heads * 48), 64, 0, s>>>(attn, p.temp);

  float* ao = ws.alloc((size_t)C * HW);
  gemm(s, attn, v, nullptr, ao, 48, HW, 48, heads,
       48 * 48, (long long)48 * HW, (long long)48 * HW, HW, false);
  // x = x + proj(ao)   (residual fused in epilogue)
  gemm(s, p.proj, ao, x, x, C, HW, C, 1, 0, 0, 0, HW, false);

  // ---- GDFN ----
  k_layernorm<<<cdiv_u(HW, 256), 256, 0, s>>>(x, p.ln2w, p.ln2b, ln, C, HW);
  float* h1 = ws.alloc((size_t)2 * hid * HW);
  float* h2 = ws.alloc((size_t)2 * hid * HW);
  gemm(s, p.pin, ln, nullptr, h1, 2 * hid, HW, C, 1, 0, 0, 0, HW, false);
  dwconv(s, h1, p.ffn_dw, h2, 2 * hid, H, W);
  long long ng = (long long)hid * HW;
  k_gelu_gate<<<cdiv_u(ng, 256), 256, 0, s>>>(h2, h1, ng, ng);
  // x = x + pout(gate)
  gemm(s, p.pout, h1, x, x, C, HW, hid, 1, 0, 0, 0, HW, false);

  ws.off = save;
}

extern "C" void kernel_launch(void* const* d_in, const int* in_sizes, int n_in,
                              void* d_out, int out_size, void* d_ws, size_t ws_size,
                              hipStream_t stream)
{
  (void)in_sizes; (void)n_in; (void)out_size; (void)ws_size;

  // ---- parse params in JAX pytree (sorted-key) flatten order ----
  // top: {'params','x'}; params: dec, downs, embed, enc, out, reduces, ref, ups
  // block: attn{proj,qkv,qkv_dw,temp}, ffn{dw,pin,pout}, ln1_b, ln1_w, ln2_b, ln2_w
  int cur = 0;
  auto nf = [&]() { return (const float*)d_in[cur++]; };

  BlockP dec0[6], dec1[6], dec2[4];
  BlockP enc0[4], enc1[6], enc2[6], enc3[8];
  BlockP refb[4];
  auto read_blocks = [&](BlockP* arr, int n, int c, int h) {
    for (int i = 0; i < n; ++i) {
      BlockP& b = arr[i];
      b.proj = nf(); b.qkv = nf(); b.qkv_dw = nf(); b.temp = nf();
      b.ffn_dw = nf(); b.pin = nf(); b.pout = nf();
      b.ln1b = nf(); b.ln1w = nf(); b.ln2b = nf(); b.ln2w = nf();
      b.c = c; b.heads = h; b.hid = (int)((double)c * 2.66);
    }
  };
  read_blocks(dec0, 6, 192, 4);
  read_blocks(dec1, 6, 96, 2);
  read_blocks(dec2, 4, 96, 1);
  const float* downs[3]; for (int i = 0; i < 3; ++i) downs[i] = nf();
  const float* embed = nf();
  read_blocks(enc0, 4, 48, 1);
  read_blocks(enc1, 6, 96, 2);
  read_blocks(enc2, 6, 192, 4);
  read_blocks(enc3, 8, 384, 8);
  const float* outw = nf();
  const float* reduces[2]; reduces[0] = nf(); reduces[1] = nf();
  read_blocks(refb, 4, 96, 1);
  const float* ups[3]; for (int i = 0; i < 3; ++i) ups[i] = nf();
  const float* xin = nf();

  Bump ws{(char*)d_ws, 0};
  hipStream_t s = stream;

  // ---- encoder ----
  float* e1 = ws.alloc((size_t)48 * 16384);                 // 48 x 128 x 128
  conv3x3(s, xin, embed, e1, 3, 48, 128, 128);
  for (int i = 0; i < 4; ++i) tblock(s, ws, enc0[i], e1, 128, 128);

  float* t1 = ws.alloc((size_t)24 * 16384);
  conv3x3(s, e1, downs[0], t1, 48, 24, 128, 128);
  float* e2 = ws.alloc((size_t)96 * 4096);                  // 96 x 64 x 64
  k_unshuffle2<<<cdiv_u((long long)96 * 4096, 256), 256, 0, s>>>(t1, e2, 24, 128, 128);
  for (int i = 0; i < 6; ++i) tblock(s, ws, enc1[i], e2, 64, 64);

  float* t2 = ws.alloc((size_t)48 * 4096);
  conv3x3(s, e2, downs[1], t2, 96, 48, 64, 64);
  float* e3 = ws.alloc((size_t)192 * 1024);                 // 192 x 32 x 32
  k_unshuffle2<<<cdiv_u((long long)192 * 1024, 256), 256, 0, s>>>(t2, e3, 48, 64, 64);
  for (int i = 0; i < 6; ++i) tblock(s, ws, enc2[i], e3, 32, 32);

  float* t3 = ws.alloc((size_t)96 * 1024);
  conv3x3(s, e3, downs[2], t3, 192, 96, 32, 32);
  float* e4 = ws.alloc((size_t)384 * 256);                  // 384 x 16 x 16
  k_unshuffle2<<<cdiv_u((long long)384 * 256, 256), 256, 0, s>>>(t3, e4, 96, 32, 32);
  for (int i = 0; i < 8; ++i) tblock(s, ws, enc3[i], e4, 16, 16);

  // ---- decoder level 3 ----
  float* t4 = ws.alloc((size_t)768 * 256);
  conv3x3(s, e4, ups[0], t4, 384, 768, 16, 16);
  float* u0 = ws.alloc((size_t)192 * 1024);
  k_shuffle2<<<cdiv_u((long long)192 * 1024, 256), 256, 0, s>>>(t4, u0, 768, 16, 16);
  float* cat3 = ws.alloc((size_t)384 * 1024);
  (void)hipMemcpyAsync(cat3, u0, (size_t)192 * 1024 * 4, hipMemcpyDeviceToDevice, s);
  (void)hipMemcpyAsync(cat3 + (size_t)192 * 1024, e3, (size_t)192 * 1024 * 4,
                       hipMemcpyDeviceToDevice, s);
  float* d3 = ws.alloc((size_t)192 * 1024);
  gemm(s, reduces[0], cat3, nullptr, d3, 192, 1024, 384, 1, 0, 0, 0, 1024, false);
  for (int i = 0; i < 6; ++i) tblock(s, ws, dec0[i], d3, 32, 32);

  // ---- decoder level 2 ----
  float* t5 = ws.alloc((size_t)384 * 1024);
  conv3x3(s, d3, ups[1], t5, 192, 384, 32, 32);
  float* u1 = ws.alloc((size_t)96 * 4096);
  k_shuffle2<<<cdiv_u((long long)96 * 4096, 256), 256, 0, s>>>(t5, u1, 384, 32, 32);
  float* cat2 = ws.alloc((size_t)192 * 4096);
  (void)hipMemcpyAsync(cat2, u1, (size_t)96 * 4096 * 4, hipMemcpyDeviceToDevice, s);
  (void)hipMemcpyAsync(cat2 + (size_t)96 * 4096, e2, (size_t)96 * 4096 * 4,
                       hipMemcpyDeviceToDevice, s);
  float* d2 = ws.alloc((size_t)96 * 4096);
  gemm(s, reduces[1], cat2, nullptr, d2, 96, 4096, 192, 1, 0, 0, 0, 4096, false);
  for (int i = 0; i < 6; ++i) tblock(s, ws, dec1[i], d2, 64, 64);

  // ---- decoder level 1 (no reduce; runs on concat directly) ----
  float* t6 = ws.alloc((size_t)192 * 4096);
  conv3x3(s, d2, ups[2], t6, 96, 192, 64, 64);
  float* u2 = ws.alloc((size_t)48 * 16384);
  k_shuffle2<<<cdiv_u((long long)48 * 16384, 256), 256, 0, s>>>(t6, u2, 192, 64, 64);
  float* fd = ws.alloc((size_t)96 * 16384);
  (void)hipMemcpyAsync(fd, u2, (size_t)48 * 16384 * 4, hipMemcpyDeviceToDevice, s);
  (void)hipMemcpyAsync(fd + (size_t)48 * 16384, e1, (size_t)48 * 16384 * 4,
                       hipMemcpyDeviceToDevice, s);
  for (int i = 0; i < 4; ++i) tblock(s, ws, dec2[i], fd, 128, 128);

  // ---- refinement ----
  for (int i = 0; i < 4; ++i) tblock(s, ws, refb[i], fd, 128, 128);

  // ---- output conv + residual ----
  float* o = ws.alloc((size_t)3 * 16384);
  conv3x3(s, fd, outw, o, 96, 3, 128, 128);
  k_add<<<cdiv_u((long long)3 * 16384, 256), 256, 0, s>>>(o, xin, (float*)d_out,
                                                          (long long)3 * 16384);
}
